// GRUModel_27504970564214
// MI455X (gfx1250) — compile-verified, matrix operand order checked by reading, not run on hardware
//
#include <hip/hip_runtime.h>
#include <cstdint>
#include <cstddef>

typedef __attribute__((ext_vector_type(16))) _Float16 v16h;
typedef __attribute__((ext_vector_type(8)))  _Float16 v8h;
typedef __attribute__((ext_vector_type(8)))  float    v8f;

#define B_   64
#define T_   4096
#define D_   32
#define H_   128
#define G3   384      // 3*H
#define KS   5        // k-steps of 32: 4 over h (K=128) + 1 over u (K=32)
#define H16S 136      // padded halfs per h_f16 row (bank-conflict-free A reads)

// fast 1-ulp reciprocal (v_rcp_f32) -- avoids IEEE fdiv expansion
__device__ __forceinline__ float frcp_(float x) { return __builtin_amdgcn_rcpf(x); }

__device__ __forceinline__ float sigmoidf_(float x) {
  return frcp_(1.0f + __expf(-x));
}
__device__ __forceinline__ float tanhf_(float x) {
  // tanh(x) = 1 - 2/(exp(2x)+1); saturates correctly at +-1
  return 1.0f - 2.0f * frcp_(1.0f + __expf(2.0f * x));
}

// Build a WMMA 16x16x32 f16 operand fragment (one lane's 16 halves) from a
// row-major f32 row:  halves j=0..7  <- row[g*8 + j]
//                     halves j=8..15 <- row[16 + g*8 + (j-8)]
// (ISA 05_wmma §7.12.2 16-bit A/B lane layout, lane-group g = lane>>4)
__device__ __forceinline__ v16h frag_from_f32row(const float* __restrict__ row, int g) {
  const float4 f0 = *(const float4*)(row + g * 8);
  const float4 f1 = *(const float4*)(row + g * 8 + 4);
  const float4 f2 = *(const float4*)(row + 16 + g * 8);
  const float4 f3 = *(const float4*)(row + 16 + g * 8 + 4);
  const float uf[16] = {f0.x, f0.y, f0.z, f0.w, f1.x, f1.y, f1.z, f1.w,
                        f2.x, f2.y, f2.z, f2.w, f3.x, f3.y, f3.z, f3.w};
  v16h r;
#pragma unroll
  for (int j = 0; j < 16; ++j) r[j] = (_Float16)uf[j];
  return r;
}

__device__ __forceinline__ v8f splat8(float x) {
  v8f r = {x, x, x, x, x, x, x, x};
  return r;
}

__global__ __launch_bounds__(256, 1)
void gru_fused_persistent(const float* __restrict__ u,
                          const float* __restrict__ w_ih,
                          const float* __restrict__ w_hh,
                          const float* __restrict__ b_ih,
                          const float* __restrict__ b_hh,
                          const float* __restrict__ fc_w,
                          const float* __restrict__ fc_b,
                          float* __restrict__ out) {
  // ---- LDS: only the per-step producer/consumer staging (~37 KB) ----
  __shared__ _Float16 h_f16[16 * H16S];  // h state, f16 (A-matrix source)
  __shared__ float    hg_s[16 * G3];     // gate pre-acts (r,z fused; hn for n)
  __shared__ float    xn_s[16 * H_];     // x-side of n gate (kept separate)

  const int tid  = threadIdx.x;
  const int b0   = blockIdx.x * 16;  // this WG's 16 batch rows (independent chains)
  const int wave = tid >> 5;         // 8 waves (wave32)
  const int lane = tid & 31;
  const int l    = lane & 15;
  const int g    = lane >> 4;

  // ---- setup: B = [w_hh^T ; w_ih^T] fragments live in VGPRs for all 4096 steps
  v16h bfr[3][KS];
  float cbh[3], cbx[3];
#pragma unroll
  for (int tt = 0; tt < 3; ++tt) {
    const int nt = wave * 3 + tt;
    const int n  = nt * 16 + l;
#pragma unroll
    for (int ks = 0; ks < 4; ++ks)
      bfr[tt][ks] = frag_from_f32row(w_hh + (size_t)n * H_ + ks * 32, g);
    bfr[tt][4] = frag_from_f32row(w_ih + (size_t)n * D_, g);
    cbh[tt] = b_hh[n];          // h-side bias -> init of ch accumulator
    cbx[tt] = b_ih[n];          // x-side bias -> init of cx accumulator
  }

  // gate-side constants in registers
  const int gb = tid >> 4;          // batch row this thread owns in gate phase
  const int k0 = (tid & 15) * 8;    // its 8 hidden elements
  float fcw_r[8];
#pragma unroll
  for (int kk = 0; kk < 8; ++kk) fcw_r[kk] = fc_w[k0 + kk];
  const float fcb = fc_b[0];
  float hp[8] = {0.f, 0.f, 0.f, 0.f, 0.f, 0.f, 0.f, 0.f};  // h state, exact f32

  for (int i = tid; i < 16 * H16S; i += 256) h_f16[i] = (_Float16)0.0f;

  // software-pipelined u fragment (k-step 4 of A): preload t=0
  v16h a4 = frag_from_f32row(u + (size_t)(b0 + l) * T_ * D_, g);
  __syncthreads();

  // ---- serial recurrence over T, fully resident in one WGP ----
  for (int t = 0; t < T_; ++t) {
    // A fragments over h (k-steps 0..3); only LDS loads on the critical path
    v16h a[4];
    {
      const _Float16* hrow = &h_f16[l * H16S];
#pragma unroll
      for (int ks = 0; ks < 4; ++ks) {
        v8h lo = *(const v8h*)(hrow + ks * 32 + g * 8);
        v8h hi = *(const v8h*)(hrow + ks * 32 + 16 + g * 8);
#pragma unroll
        for (int j = 0; j < 8; ++j) { a[ks][j] = lo[j]; a[ks][8 + j] = hi[j]; }
      }
    }

    // 3 N-tiles per wave, uniform wmma code. Biases ride in the C initializers
    // (splat movs overlap the ds_load waits); split tiles store wmma D directly.
#pragma unroll
    for (int tt = 0; tt < 3; ++tt) {
      v8f cx = splat8(cbx[tt]);   // x-part: ready first (a4 already in regs)
      cx = __builtin_amdgcn_wmma_f32_16x16x32_f16(false, a4, false, bfr[tt][4],
                                                  (short)0, cx, false, false);
      v8f ch = splat8(cbh[tt]);   // h-part over k-steps 0..3
#pragma unroll
      for (int ks = 0; ks < 4; ++ks)
        ch = __builtin_amdgcn_wmma_f32_16x16x32_f16(false, a[ks], false, bfr[tt][ks],
                                                    (short)0, ch, false, false);

      const int nt = wave * 3 + tt;
      const int n  = nt * 16 + l;
      if (nt < 16) {
        // r,z gates: x+h parts summed (biases already inside)
#pragma unroll
        for (int r = 0; r < 8; ++r)
          hg_s[(r + 8 * g) * G3 + n] = ch[r] + cx[r];
      } else {
        // n gate: hn and xn must stay separate (n = tanh(xn + r*hn))
#pragma unroll
        for (int r = 0; r < 8; ++r)
          hg_s[(r + 8 * g) * G3 + n] = ch[r];
#pragma unroll
        for (int r = 0; r < 8; ++r)
          xn_s[(r + 8 * g) * H_ + (n - 2 * H_)] = cx[r];
      }
    }
    __syncthreads();

    // ---- gate phase: also prefetch + pipeline u for t+1 (off critical path)
    if (tid < 16 && t + 8 < T_)
      __builtin_prefetch(u + ((size_t)(b0 + tid) * T_ + (t + 8)) * D_, 0, 0);
    v16h a4n = a4;
    if (t + 1 < T_)
      a4n = frag_from_f32row(u + ((size_t)(b0 + l) * T_ + (t + 1)) * D_, g);

    {
      float acc = 0.0f;
#pragma unroll
      for (int kk = 0; kk < 8; ++kk) {
        const int k = k0 + kk;
        const float rr = sigmoidf_(hg_s[gb * G3 + k]);
        const float zz = sigmoidf_(hg_s[gb * G3 + H_ + k]);
        const float hn = hg_s[gb * G3 + 2 * H_ + k];
        const float xn = xn_s[gb * H_ + k];
        const float nn = tanhf_(xn + rr * hn);
        const float hv = (1.0f - zz) * nn + zz * hp[kk];
        hp[kk] = hv;                           // exact f32 state in registers
        h_f16[gb * H16S + k] = (_Float16)hv;   // f16 copy for next step's A
        acc += hv * fcw_r[kk];
      }
      // FC head: reduce across the 16 lanes sharing this batch row
      acc += __shfl_xor(acc, 1);
      acc += __shfl_xor(acc, 2);
      acc += __shfl_xor(acc, 4);
      acc += __shfl_xor(acc, 8);
      if ((tid & 15) == 0) out[(size_t)(b0 + gb) * T_ + t] = acc + fcb;
    }
    a4 = a4n;
    __syncthreads();
  }
}

extern "C" void kernel_launch(void* const* d_in, const int* in_sizes, int n_in,
                              void* d_out, int out_size, void* d_ws, size_t ws_size,
                              hipStream_t stream) {
  (void)in_sizes; (void)n_in; (void)out_size; (void)d_ws; (void)ws_size;
  const float* u    = (const float*)d_in[0];
  const float* w_ih = (const float*)d_in[1];
  const float* w_hh = (const float*)d_in[2];
  const float* b_ih = (const float*)d_in[3];
  const float* b_hh = (const float*)d_in[4];
  const float* fc_w = (const float*)d_in[5];
  const float* fc_b = (const float*)d_in[6];
  float* out = (float*)d_out;
  // 4 independent batch-slices of 16 rows; each WG is a persistent GRU chain.
  gru_fused_persistent<<<4, 256, 0, stream>>>(u, w_ih, w_hh, b_ih, b_hh,
                                              fc_w, fc_b, out);
}
